// RNN_87668872446413
// MI455X (gfx1250) — compile-verified, matrix operand order checked by reading, not run on hardware
//
#include <hip/hip_runtime.h>
#include <hip/hip_bf16.h>
#include <math.h>

// ---------------------------------------------------------------------------
// CDNA5 (gfx1250) RNN-LM forward: bf16 WMMA GEMMs (fp32 accumulate),
// async global->LDS A-tile staging (ASYNCcnt), streaming softmax NLL.
// ---------------------------------------------------------------------------

typedef __attribute__((ext_vector_type(16))) __bf16 v16bf;
typedef __attribute__((ext_vector_type(8)))  float  v8f;

union FragB16 {
    v16bf v;
    uint4 q[2];
};

__device__ __forceinline__ unsigned short f2bf(float f) {
    unsigned int u = __float_as_uint(f);
    unsigned int r = u + 0x7FFFu + ((u >> 16) & 1u);   // round-to-nearest-even
    return (unsigned short)(r >> 16);
}

// Load one 16x32 bf16 A/B fragment from a row-major [rows][ld] bf16 matrix.
// CDNA5 16-bit 16x32 layout: lanes 0-15 hold K 0..7 & 16..23, lanes 16-31
// hold K 8..15 & 24..31, all for row (base + lane&15).
__device__ __forceinline__ FragB16 load_frag_g(const unsigned short* __restrict__ base,
                                               int row0, int ld, int k0, int lane) {
    const int half = lane >> 4;
    const int r    = lane & 15;
    FragB16 f;
    const unsigned short* p = base + (size_t)(row0 + r) * ld + k0 + 8 * half;
    f.q[0] = *(const uint4*)(p);        // K = k0 + 8*half .. +7
    f.q[1] = *(const uint4*)(p + 16);   // K = k0 + 16 + 8*half .. +7
    return f;
}

// ---------------------------------------------------------------------------
// bf16 conversion / gather kernels (one-time per launch)
// ---------------------------------------------------------------------------

// X[127][32][320] <- bf16(wordvec[sent[b][t]][k]), zero-padded K 300->320
__global__ void embed_kernel(const int* __restrict__ sent,
                             const float* __restrict__ wordvec,
                             unsigned short* __restrict__ X) {
    int i = blockIdx.x * blockDim.x + threadIdx.x;
    if (i >= 127 * 32 * 320) return;
    int k = i % 320;
    int rest = i / 320;
    int b = rest % 32;
    int t = rest / 32;
    unsigned short v = 0;
    if (k < 300) {
        int tok = sent[b * 128 + t];
        v = f2bf(wordvec[(size_t)tok * 300 + k]);
    }
    X[i] = v;
}

// WxhT[1024][320] <- bf16(Wxh[k][n]), zero-padded K
__global__ void cvt_wxhT(const float* __restrict__ Wxh, unsigned short* __restrict__ WT) {
    int i = blockIdx.x * blockDim.x + threadIdx.x;
    if (i >= 1024 * 320) return;
    int k = i % 320, n = i / 320;
    WT[i] = (k < 300) ? f2bf(Wxh[(size_t)k * 1024 + n]) : (unsigned short)0;
}

// WhhT[1024][1024] <- bf16(Whh[k][n])
__global__ void cvt_whhT(const float* __restrict__ Whh, unsigned short* __restrict__ WT) {
    int i = blockIdx.x * blockDim.x + threadIdx.x;
    if (i >= 1024 * 1024) return;
    int k = i % 1024, n = i / 1024;
    WT[i] = f2bf(Whh[(size_t)k * 1024 + n]);
}

// WoutT[32000][1024] <- bf16(Wout[h][v])
__global__ void cvt_woutT(const float* __restrict__ Wout, unsigned short* __restrict__ WT) {
    long long i = (long long)blockIdx.x * blockDim.x + threadIdx.x;
    if (i >= 32000LL * 1024) return;
    int h = (int)(i % 1024);
    int v = (int)(i / 1024);
    WT[i] = f2bf(Wout[(size_t)h * 32000 + v]);
}

// ---------------------------------------------------------------------------
// Recurrent step: Hout[32][1024] = tanh(Xt@Wxh + Hprev@Whh + bh), bf16 WMMA.
// 128 wave-tiles (2 M x 64 N), 32 blocks x 4 waves.
// ---------------------------------------------------------------------------
__global__ void __launch_bounds__(128) rnn_step_kernel(
    const unsigned short* __restrict__ Xt,     // [32][320] bf16
    const unsigned short* __restrict__ Hprev,  // [32][1024] bf16
    const unsigned short* __restrict__ WxhT,   // [1024][320] bf16
    const unsigned short* __restrict__ WhhT,   // [1024][1024] bf16
    const float* __restrict__ bh,
    unsigned short* __restrict__ Hout,         // [32][1024] bf16
    int has_prev)
{
    const int lane  = threadIdx.x & 31;
    const int wave  = threadIdx.x >> 5;
    const int tile  = blockIdx.x * 4 + wave;   // 0..127
    const int m0    = (tile & 1) * 16;
    const int n0    = (tile >> 1) * 16;
    const int half  = lane >> 4;
    const int r     = lane & 15;

    v8f acc = {};
    #pragma unroll 2
    for (int kt = 0; kt < 320; kt += 32) {
        FragB16 a = load_frag_g(Xt,   m0, 320, kt, lane);
        FragB16 b = load_frag_g(WxhT, n0, 320, kt, lane);
        acc = __builtin_amdgcn_wmma_f32_16x16x32_bf16(false, a.v, false, b.v,
                                                      (short)0, acc, false, false);
    }
    if (has_prev) {
        #pragma unroll 2
        for (int kt = 0; kt < 1024; kt += 32) {
            FragB16 a = load_frag_g(Hprev, m0, 1024, kt, lane);
            FragB16 b = load_frag_g(WhhT,  n0, 1024, kt, lane);
            acc = __builtin_amdgcn_wmma_f32_16x16x32_bf16(false, a.v, false, b.v,
                                                          (short)0, acc, false, false);
        }
    }
    const int n = n0 + r;
    const float bias = bh[n];
    #pragma unroll
    for (int j = 0; j < 8; ++j) {
        int row = m0 + half * 8 + j;           // batch index
        float h = tanhf(acc[j] + bias);
        Hout[(size_t)row * 1024 + n] = f2bf(h);
    }
}

// ---------------------------------------------------------------------------
// Output projection: logits[b][s][v] = hs[s][b][:] . WoutT[v][:] + bout[v].
// hs flat is row-major [4064][1024] with row = s*32+b.
// Block = 256 thr (8 waves). 32-row x 1024-K A tile (64 KB) staged into LDS
// via global_load_async_to_lds_b128 (ASYNCcnt path). Each wave computes a
// 32x16 output (two M-tiles sharing one B fragment) -> halves B traffic.
// Grid = (32000/128, 4064/32) = (250, 127).
// ---------------------------------------------------------------------------
__global__ void __launch_bounds__(256) logits_kernel(
    const unsigned short* __restrict__ hs,     // [4064][1024] bf16
    const unsigned short* __restrict__ WoutT,  // [32000][1024] bf16
    const float* __restrict__ bout,
    float* __restrict__ out)                   // [32][127][32000] fp32
{
    __shared__ __align__(16) unsigned short As[32 * 1024];   // 64 KB

    const int m0 = blockIdx.y * 32;

    // Async stage of the 32x1024 bf16 A tile: 4096 x B128 transfers,
    // 16 per thread, tracked on ASYNCcnt (no VGPR round-trip).
    {
        const char* gsrc = (const char*)(hs + (size_t)m0 * 1024);
        unsigned lds0 = (unsigned)(unsigned long long)(void*)As;
        #pragma unroll
        for (int it = 0; it < 16; ++it) {
            unsigned idx  = threadIdx.x + 256u * it;     // uint4 index
            unsigned loff = lds0 + idx * 16u;
            const char* ga = gsrc + (size_t)idx * 16u;
            asm volatile("global_load_async_to_lds_b128 %0, %1, off"
                         :: "v"(loff), "v"(ga) : "memory");
        }
        asm volatile("s_wait_asynccnt 0x0" ::: "memory");
    }
    __syncthreads();

    const int lane = threadIdx.x & 31;
    const int wave = threadIdx.x >> 5;
    const int n0   = (blockIdx.x * 8 + wave) * 16;
    const int half = lane >> 4;
    const int r    = lane & 15;

    const unsigned short* browp = WoutT + (size_t)(n0 + r) * 1024;

    v8f acc0 = {};   // rows m0 .. m0+15
    v8f acc1 = {};   // rows m0+16 .. m0+31
    #pragma unroll 4
    for (int kt = 0; kt < 1024; kt += 32) {
        FragB16 a0, a1, b;
        const unsigned short* pa0 = As + r * 1024 + kt + 8 * half;          // LDS
        const unsigned short* pa1 = As + (16 + r) * 1024 + kt + 8 * half;   // LDS
        a0.q[0] = *(const uint4*)(pa0);
        a0.q[1] = *(const uint4*)(pa0 + 16);
        a1.q[0] = *(const uint4*)(pa1);
        a1.q[1] = *(const uint4*)(pa1 + 16);
        const unsigned short* pb = browp + kt + 8 * half;
        __builtin_prefetch(pb + 128, 0, 0);              // next K-chunk of B
        b.q[0] = *(const uint4*)(pb);
        b.q[1] = *(const uint4*)(pb + 16);
        acc0 = __builtin_amdgcn_wmma_f32_16x16x32_bf16(false, a0.v, false, b.v,
                                                       (short)0, acc0, false, false);
        acc1 = __builtin_amdgcn_wmma_f32_16x16x32_bf16(false, a1.v, false, b.v,
                                                       (short)0, acc1, false, false);
    }

    const int n = n0 + r;
    const float bias = bout[n];
    #pragma unroll
    for (int j = 0; j < 8; ++j) {
        int g0 = m0 + half * 8 + j;            // = s*32 + b   (M-tile 0)
        int g1 = g0 + 16;                      //              (M-tile 1)
        out[((size_t)(g0 & 31) * 127 + (g0 >> 5)) * 32000 + n] = acc0[j] + bias;
        out[((size_t)(g1 & 31) * 127 + (g1 >> 5)) * 32000 + n] = acc1[j] + bias;
    }
}

// ---------------------------------------------------------------------------
// Masked NLL: per (b,s) row, max + logsumexp over 32000, atomic accumulate.
// ---------------------------------------------------------------------------
__global__ void __launch_bounds__(256) nll_kernel(
    const float* __restrict__ logits,   // [32][127][32000]
    const int* __restrict__ sent,
    const int* __restrict__ length,
    float* __restrict__ accum)
{
    __shared__ float red[256];
    const int bs = blockIdx.x;          // b*127 + s
    const int b = bs / 127;
    const int s = bs % 127;
    const float* row = logits + (size_t)bs * 32000;

    float m = -INFINITY;
    for (int v = threadIdx.x; v < 32000; v += 256) m = fmaxf(m, row[v]);
    red[threadIdx.x] = m;
    __syncthreads();
    for (int st = 128; st > 0; st >>= 1) {
        if (threadIdx.x < st)
            red[threadIdx.x] = fmaxf(red[threadIdx.x], red[threadIdx.x + st]);
        __syncthreads();
    }
    m = red[0];
    __syncthreads();

    float sum = 0.f;
    for (int v = threadIdx.x; v < 32000; v += 256) sum += __expf(row[v] - m);
    red[threadIdx.x] = sum;
    __syncthreads();
    for (int st = 128; st > 0; st >>= 1) {
        if (threadIdx.x < st) red[threadIdx.x] += red[threadIdx.x + st];
        __syncthreads();
    }

    if (threadIdx.x == 0 && s < length[b]) {
        int tgt = sent[b * 128 + (s + 1)];
        float nll = m + __logf(red[0]) - row[tgt];
        atomicAdd(accum, nll);
    }
}

__global__ void init_accum(float* accum) {
    if (threadIdx.x == 0) accum[0] = 0.f;
}

__global__ void finalize_kernel(const float* __restrict__ accum,
                                const int* __restrict__ length,
                                float* __restrict__ out0) {
    if (threadIdx.x == 0) {
        float cnt = 0.f;
        for (int b = 0; b < 32; ++b) cnt += (float)length[b];
        out0[0] = accum[0] / cnt;
    }
}

// ---------------------------------------------------------------------------
extern "C" void kernel_launch(void* const* d_in, const int* in_sizes, int n_in,
                              void* d_out, int out_size, void* d_ws, size_t ws_size,
                              hipStream_t stream)
{
    const int*   sent    = (const int*)d_in[0];
    const int*   length  = (const int*)d_in[1];
    const float* wordvec = (const float*)d_in[2];
    const float* Wxh     = (const float*)d_in[3];
    const float* Whh     = (const float*)d_in[4];
    const float* bh      = (const float*)d_in[5];
    const float* Wout    = (const float*)d_in[6];
    const float* bout    = (const float*)d_in[7];

    float* out    = (float*)d_out;
    float* logits = out + 1;   // d_out = [loss, logits(32*127*32000)]

    char* ws = (char*)d_ws;
    size_t off = 0;
    auto alloc = [&](size_t bytes) -> void* {
        void* p = ws + off;
        off += (bytes + 255) & ~(size_t)255;
        return p;
    };
    unsigned short* X     = (unsigned short*)alloc(127ull * 32 * 320 * 2);   // 2.6 MB
    unsigned short* WxhT  = (unsigned short*)alloc(1024ull * 320 * 2);
    unsigned short* WhhT  = (unsigned short*)alloc(1024ull * 1024 * 2);
    unsigned short* WoutT = (unsigned short*)alloc(32000ull * 1024 * 2);     // 65.5 MB
    unsigned short* hsb   = (unsigned short*)alloc(127ull * 32 * 1024 * 2);  // 8.3 MB
    float* accum          = (float*)alloc(256);
    (void)ws_size; (void)in_sizes; (void)n_in; (void)out_size;

    { int n = 127 * 32 * 320; embed_kernel<<<(n + 255) / 256, 256, 0, stream>>>(sent, wordvec, X); }
    { int n = 1024 * 320;     cvt_wxhT<<<(n + 255) / 256, 256, 0, stream>>>(Wxh, WxhT); }
    { int n = 1024 * 1024;    cvt_whhT<<<(n + 255) / 256, 256, 0, stream>>>(Whh, WhhT); }
    { long long n = 32000LL * 1024; cvt_woutT<<<(unsigned)((n + 255) / 256), 256, 0, stream>>>(Wout, WoutT); }
    init_accum<<<1, 32, 0, stream>>>(accum);

    // sequential recurrence: 127 dependent step kernels
    for (int t = 0; t < 127; ++t) {
        const unsigned short* Xt    = X   + (size_t)t * 32 * 320;
        const unsigned short* Hprev = hsb + (size_t)(t > 0 ? t - 1 : 0) * 32 * 1024;
        unsigned short*       Hout  = hsb + (size_t)t * 32 * 1024;
        rnn_step_kernel<<<32, 128, 0, stream>>>(Xt, Hprev, WxhT, WhhT, bh, Hout, t > 0);
    }

    // big projection GEMM: (4064 x 1024) @ (1024 x 32000)
    dim3 gg(250, 127);
    logits_kernel<<<gg, 256, 0, stream>>>(hsb, WoutT, bout, logits);

    nll_kernel<<<32 * 127, 256, 0, stream>>>(logits, sent, length, accum);
    finalize_kernel<<<1, 32, 0, stream>>>(accum, length, out);
}
